// PointMLP_12386685681898
// MI455X (gfx1250) — matrix operand order repaired and checked by reading, not compiled, not run on hardware
//
#include <hip/hip_runtime.h>
#include <hip/hip_bf16.h>

typedef __attribute__((ext_vector_type(16))) _Float16 v16h;
typedef __attribute__((ext_vector_type(8)))  _Float16 v8h;
typedef __attribute__((ext_vector_type(8)))  float    v8f;

#define BN_SC 0.999995000037f  // 1/sqrt(1+1e-5)

// ---------------------------------------------------------------------------
// WMMA GEMM: out[m, o] = epilogue( sum_k X[m,k] * W[o,k] )
// X: f16 [M, Kp] row-major (Kp multiple of 32, zero padded)
// W: f16 [Cout, Kp] row-major (zero padded)
//
// Block = 256 threads = 8 waves arranged 4(M) x 2(N).
// Block tile: 128 rows x 128 cols. Wave tile: 32 rows x 64 cols
//   -> 2 A-frags x 4 B-frags = 8 v_wmma_f32_16x16x32_f16 per K-step.
// A tile (128x32 f16 = 8KB) is staged into LDS with CDNA5 async copies
// (global_load_async_to_lds_b128, ASYNCcnt), double buffered so the next
// K-step's copy overlaps the current K-step's WMMAs. B fragments are direct
// global_load_b128 (weights are L2-resident; 4 M-waves share the same rows).
// Epilogue: v = (acc + bias[col]) * scale; v += res; relu; write f16/f32.
// ---------------------------------------------------------------------------
__global__ __launch_bounds__(256) void k_gemm(
    const _Float16* __restrict__ X, int Kp,
    const _Float16* __restrict__ W,
    const float* __restrict__ bias,
    int M, int Cout, int Np,
    float scale, int relu,
    const _Float16* __restrict__ res,
    _Float16* __restrict__ outH,
    float* __restrict__ outF, int ldF)
{
    __shared__ _Float16 lx[2][128 * 32];   // 2 x 8KB A tiles

    const int lane  = threadIdx.x & 31;
    const int wave  = threadIdx.x >> 5;
    const int waveM = wave & 3;            // 0..3  (M direction)
    const int waveN = wave >> 2;           // 0..1  (N direction)
    const int m     = lane & 15;
    const int half  = lane >> 4;

    const long blockRow = (long)blockIdx.x * 128;
    const int  colBase  = blockIdx.y * 128 + waveN * 64;

    // B row pointers (clamped; stores are guarded)
    const _Float16* wrow[4];
#pragma unroll
    for (int t = 0; t < 4; ++t) {
        int c = colBase + 16 * t + m;
        if (c >= Cout) c = Cout - 1;
        if (c < 0) c = 0;
        wrow[t] = W + (long)c * Kp;
    }

    // local A rows for this wave's two 16-row fragments
    const int aRow0 = waveM * 32 + m;
    const int aRow1 = waveM * 32 + 16 + m;

    v8f acc[2][4];
#pragma unroll
    for (int f = 0; f < 2; ++f)
#pragma unroll
        for (int t = 0; t < 4; ++t)
#pragma unroll
            for (int r = 0; r < 8; ++r) acc[f][t][r] = 0.0f;

    // async copy of one 128x32 A tile into LDS buffer `buf`
    auto copyA = [&](int buf, int k0) {
#pragma unroll
        for (int c = (int)threadIdx.x; c < 512; c += 256) {
            int row = c >> 2;        // 0..127
            int o16 = c & 3;         // 16B chunk within 64B row
            long gr = blockRow + row;
            if (gr >= M) gr = M - 1;
            const _Float16* gp = X + gr * (long)Kp + k0 + o16 * 8;
            unsigned la = (unsigned)(uintptr_t)&lx[buf][row * 32 + o16 * 8];
            asm volatile("global_load_async_to_lds_b128 %0, %1, off"
                         :: "v"(la), "v"(gp) : "memory");
        }
    };

    const int nk = Kp >> 5;
    copyA(0, 0);

    for (int s = 0; s < nk; ++s) {
        const int cur = s & 1;
        const int k0  = s << 5;
        if (s + 1 < nk) {
            copyA(cur ^ 1, (s + 1) << 5);
            asm volatile("s_wait_asynccnt 0x2" ::: "memory");
        } else {
            asm volatile("s_wait_asynccnt 0x0" ::: "memory");
        }
        __syncthreads();   // LDS tile `cur` ready for all waves

        // A fragments from LDS (16-bit A 16x32 layout):
        //   a[0..7]  = X[m][k0 + 8*half .. +8) ; a[8..15] = X[m][k0+16+8*half ..)
        v16h A[2];
        {
            v8h a0 = *(const v8h*)&lx[cur][aRow0 * 32 + 8 * half];
            v8h a1 = *(const v8h*)&lx[cur][aRow0 * 32 + 16 + 8 * half];
#pragma unroll
            for (int j = 0; j < 8; ++j) { A[0][j] = a0[j]; A[0][8 + j] = a1[j]; }
            v8h a2 = *(const v8h*)&lx[cur][aRow1 * 32 + 8 * half];
            v8h a3 = *(const v8h*)&lx[cur][aRow1 * 32 + 16 + 8 * half];
#pragma unroll
            for (int j = 0; j < 8; ++j) { A[1][j] = a2[j]; A[1][8 + j] = a3[j]; }
        }
        // B fragments (32x16): b[0..7] = Bmat[8*half + j][n] ; b[8..15] = Bmat[16 + 8*half + j][n]
#pragma unroll
        for (int t = 0; t < 4; ++t) {
            v8h b0 = *(const v8h*)(wrow[t] + k0 + 8 * half);
            v8h b1 = *(const v8h*)(wrow[t] + k0 + 16 + 8 * half);
            v16h Bf;
#pragma unroll
            for (int j = 0; j < 8; ++j) { Bf[j] = b0[j]; Bf[8 + j] = b1[j]; }
#pragma unroll
            for (int f = 0; f < 2; ++f) {
                acc[f][t] = __builtin_amdgcn_wmma_f32_16x16x32_f16(
                    false, A[f], false, Bf, (short)0, acc[f][t], false, false);
            }
        }
        __syncthreads();   // all reads of tile `cur` done before it is rewritten
    }

    // C/D layout: VGPR r: M = r + 8*half, N = lane&15
#pragma unroll
    for (int f = 0; f < 2; ++f) {
        const long fRow = blockRow + waveM * 32 + f * 16;
#pragma unroll
        for (int t = 0; t < 4; ++t) {
            const int col = colBase + 16 * t + m;
#pragma unroll
            for (int r = 0; r < 8; ++r) {
                const long orow = fRow + r + 8 * half;
                if (orow >= M) continue;
                if (col < Cout) {
                    float v = (acc[f][t][r] + bias[col]) * scale;
                    if (res)  v += (float)res[orow * (long)Np + col];
                    if (relu && v < 0.0f) v = 0.0f;
                    if (outH) outH[orow * (long)Np + col] = (_Float16)v;
                    if (outF) outF[orow * (long)ldF + col] = v;
                } else if (col < Np && outH) {
                    outH[orow * (long)Np + col] = (_Float16)0.0f;
                }
            }
        }
    }
}

// --------------------------- utility kernels -------------------------------
__global__ void k_cvt_pad(const float* __restrict__ src, _Float16* __restrict__ dst,
                          int rows, int cols, int colsPad, long total) {
    long g = (long)blockIdx.x * 256 + threadIdx.x;
    if (g >= total) return;
    int  c = (int)(g % colsPad);
    long r = g / colsPad;
    dst[g] = (c < cols) ? (_Float16)src[r * (long)cols + c] : (_Float16)0.0f;
}

__global__ void k_fill16(_Float16* p, long n) {
    long g = (long)blockIdx.x * 256 + threadIdx.x;
    if (g < n) p[g] = (_Float16)0.0f;
}

__global__ void k_zero_f32(float* p, int n) {
    int g = blockIdx.x * 256 + threadIdx.x;
    if (g < n) p[g] = 0.0f;
}

// --------------------------- FPS -------------------------------------------
__global__ void k_fps(const float* __restrict__ xyz, int Npt, int np,
                      int* __restrict__ fidx, float* __restrict__ distbuf) {
    int b = blockIdx.x;
    const float* P = xyz + (long)b * Npt * 3;
    float* dist = distbuf + (long)b * Npt;
    __shared__ float smax[256];
    __shared__ int   simax[256];
    __shared__ int   s_far;
    int t = threadIdx.x;
    for (int i = t; i < Npt; i += 256) dist[i] = 1e10f;
    if (t == 0) s_far = 0;
    __syncthreads();
    for (int it = 0; it < np; ++it) {
        int far = s_far;
        if (t == 0) fidx[b * np + it] = far;
        float cx = P[far * 3], cy = P[far * 3 + 1], cz = P[far * 3 + 2];
        float bmax = -1.0f; int bidx = 0;
        for (int i = t; i < Npt; i += 256) {
            float dx = P[i * 3] - cx, dy = P[i * 3 + 1] - cy, dz = P[i * 3 + 2] - cz;
            float d = dx * dx + dy * dy + dz * dz;
            float dm = dist[i];
            if (d < dm) { dm = d; dist[i] = d; }
            if (dm > bmax) { bmax = dm; bidx = i; }
        }
        smax[t] = bmax; simax[t] = bidx;
        __syncthreads();
        for (int s = 128; s > 0; s >>= 1) {
            if (t < s) {
                if (smax[t + s] > smax[t] ||
                    (smax[t + s] == smax[t] && simax[t + s] < simax[t])) {
                    smax[t] = smax[t + s]; simax[t] = simax[t + s];
                }
            }
            __syncthreads();
        }
        if (t == 0) s_far = simax[0];
        __syncthreads();
    }
}

__global__ void k_gather_xyz(const float* __restrict__ src, const int* __restrict__ fidx,
                             int n, int Nprev, float* __restrict__ dst, int B) {
    int g = blockIdx.x * 256 + threadIdx.x;
    if (g >= B * n) return;
    int b = g / n;
    int id = fidx[g];
    const float* s = src + ((long)b * Nprev + id) * 3;
    dst[(long)g * 3 + 0] = s[0];
    dst[(long)g * 3 + 1] = s[1];
    dst[(long)g * 3 + 2] = s[2];
}

// --------------------------- kNN (top-K smallest sq-dist) ------------------
template <int K>
__global__ void k_knn(const float* __restrict__ q, const float* __restrict__ ref,
                      int nq, int nr, int* __restrict__ oidx,
                      float* __restrict__ odist, int total) {
    int g = blockIdx.x * 128 + threadIdx.x;
    if (g >= total) return;
    int b = g / nq;
    float qx = q[(long)g * 3], qy = q[(long)g * 3 + 1], qz = q[(long)g * 3 + 2];
    const float* R = ref + (long)b * nr * 3;
    float bd[K]; int bi[K];
#pragma unroll
    for (int k = 0; k < K; ++k) { bd[k] = 3.4e38f; bi[k] = 0; }
    for (int r = 0; r < nr; ++r) {
        float dx = R[r * 3] - qx, dy = R[r * 3 + 1] - qy, dz = R[r * 3 + 2] - qz;
        float d = dx * dx + dy * dy + dz * dz;
        if (d < bd[K - 1]) {
            int p = K - 1;
            while (p > 0 && bd[p - 1] > d) { bd[p] = bd[p - 1]; bi[p] = bi[p - 1]; --p; }
            bd[p] = d; bi[p] = r;
        }
    }
    for (int k = 0; k < K; ++k) {
        oidx[(long)g * K + k] = bi[k];
        if (odist) odist[(long)g * K + k] = bd[k];
    }
}

// --------------------------- grouping: stats + normalize -------------------
__global__ void k_group_stats(const _Float16* __restrict__ feat, int ldf,
                              const float* __restrict__ xyz,
                              const int* __restrict__ fidx, const int* __restrict__ knn,
                              int n, int Nprev, int cin, int blocksPerBatch,
                              float* __restrict__ stats) {
    int b = blockIdx.x / blocksPerBatch;
    int local = (blockIdx.x % blocksPerBatch) * 256 + threadIdx.x;
    int i = local >> 5, j = local & 31;
    long qi = (long)b * n + i;
    int fid = fidx[qi];
    int id  = knn[qi * 32 + j];
    const _Float16* fa = feat + ((long)b * Nprev + id) * ldf;
    const _Float16* fm = feat + ((long)b * Nprev + fid) * ldf;
    float s = 0.0f, ss = 0.0f;
    for (int ch = 0; ch < cin; ++ch) {
        float d = (float)fa[ch] - (float)fm[ch];
        s += d; ss += d * d;
    }
    const float* xa = xyz + ((long)b * Nprev + id) * 3;
    const float* xm = xyz + ((long)b * Nprev + fid) * 3;
    for (int t = 0; t < 3; ++t) {
        float d = xa[t] - xm[t];
        s += d; ss += d * d;
    }
    __shared__ float sh1[256], sh2[256];
    sh1[threadIdx.x] = s; sh2[threadIdx.x] = ss;
    __syncthreads();
    for (int st = 128; st > 0; st >>= 1) {
        if (threadIdx.x < st) {
            sh1[threadIdx.x] += sh1[threadIdx.x + st];
            sh2[threadIdx.x] += sh2[threadIdx.x + st];
        }
        __syncthreads();
    }
    if (threadIdx.x == 0) {
        atomicAdd(&stats[b * 2 + 0], sh1[0]);
        atomicAdd(&stats[b * 2 + 1], sh2[0]);
    }
}

__global__ void k_std(const float* __restrict__ stats, float* __restrict__ stds,
                      float cnt, int B) {
    int b = threadIdx.x;
    if (b >= B) return;
    float s = stats[b * 2], ss = stats[b * 2 + 1];
    float mean = s / cnt;
    float var = (ss - s * mean) / (cnt - 1.0f);
    if (var < 0.0f) var = 0.0f;
    stds[b] = sqrtf(var);
}

__global__ void k_group_write(const _Float16* __restrict__ feat, int ldf,
                              const float* __restrict__ xyz,
                              const int* __restrict__ fidx, const int* __restrict__ knn,
                              const float* __restrict__ alpha, const float* __restrict__ beta,
                              const float* __restrict__ stds,
                              int n, int Nprev, int cin, int Kp,
                              _Float16* __restrict__ G, int total) {
    int g = blockIdx.x * 256 + threadIdx.x;   // g = ((b*n + i)*32 + j)
    if (g >= total) return;
    int nk = n * 32;
    int b = g / nk; int rem = g % nk;
    int i = rem >> 5;
    long qi = (long)b * n + i;
    int fid = fidx[qi];
    int id  = knn[(long)g];           // knn layout == grouped layout
    float inv = 1.0f / (stds[b] + 1e-5f);
    _Float16* row = G + (long)g * Kp;
    const _Float16* fa = feat + ((long)b * Nprev + id) * ldf;
    const _Float16* fm = feat + ((long)b * Nprev + fid) * ldf;
    for (int ch = 0; ch < cin; ++ch) {
        float d = (float)fa[ch] - (float)fm[ch];
        row[ch] = (_Float16)(alpha[ch] * (d * inv) + beta[ch]);
    }
    const float* xa = xyz + ((long)b * Nprev + id) * 3;
    const float* xm = xyz + ((long)b * Nprev + fid) * 3;
    for (int t = 0; t < 3; ++t) {
        float d = xa[t] - xm[t];
        row[cin + t] = (_Float16)(alpha[cin + t] * (d * inv) + beta[cin + t]);
    }
    for (int ch = 0; ch < cin; ++ch) row[cin + 3 + ch] = fm[ch];  // anchors
}

// --------------------------- pooling ---------------------------------------
__global__ void k_pool(const _Float16* __restrict__ in, int ld,
                       _Float16* __restrict__ out, long rows, int c) {
    long g = (long)blockIdx.x * 256 + threadIdx.x;
    if (g >= rows * c) return;
    int  ch = (int)(g % c);
    long r  = g / c;
    const _Float16* p = in + r * 32 * (long)ld + ch;
    float mv = (float)p[0];
    for (int j = 1; j < 32; ++j) {
        float v = (float)p[(long)j * ld];
        if (v > mv) mv = v;
    }
    out[r * (long)c + ch] = (_Float16)mv;
}

__global__ void k_maxpoints(const _Float16* __restrict__ in, int npts,
                            _Float16* __restrict__ dst, int ldD, int colOff, int B) {
    int g = blockIdx.x * 256 + threadIdx.x;
    if (g >= B * 64) return;
    int b = g / 64, ch = g % 64;
    const _Float16* p = in + (long)b * npts * 64 + ch;
    float mv = (float)p[0];
    for (int i = 1; i < npts; ++i) {
        float v = (float)p[(long)i * 64];
        if (v > mv) mv = v;
    }
    dst[(long)b * ldD + colOff + ch] = (_Float16)mv;
}

// --------------------------- concat helpers --------------------------------
__global__ void k_copy_cols(const _Float16* __restrict__ src, int ldS,
                            _Float16* __restrict__ dst, int ldD,
                            long rows, int cols, int colOff) {
    long g = (long)blockIdx.x * 256 + threadIdx.x;
    if (g >= rows * cols) return;
    int  c = (int)(g % cols);
    long r = g / cols;
    dst[r * (long)ldD + colOff + c] = src[r * (long)ldS + c];
}

__global__ void k_interp(const _Float16* __restrict__ p2, int ldp2,
                         const int* __restrict__ idx3, const float* __restrict__ d3,
                         int n1, int n2, int c2,
                         _Float16* __restrict__ dst, int ldD, int colOff, long total) {
    long g = (long)blockIdx.x * 256 + threadIdx.x;
    if (g >= total) return;
    int  ch = (int)(g % c2);
    long pi = g / c2;
    int b = (int)(pi / n1);
    const int*   id = idx3 + pi * 3;
    const float* dd = d3 + pi * 3;
    float w0 = 1.0f / (dd[0] + 1e-8f);
    float w1 = 1.0f / (dd[1] + 1e-8f);
    float w2 = 1.0f / (dd[2] + 1e-8f);
    float wsum = w0 + w1 + w2;
    float v = (w0 * (float)p2[((long)b * n2 + id[0]) * ldp2 + ch] +
               w1 * (float)p2[((long)b * n2 + id[1]) * ldp2 + ch] +
               w2 * (float)p2[((long)b * n2 + id[2]) * ldp2 + ch]) / wsum;
    dst[pi * (long)ldD + colOff + ch] = (_Float16)v;
}

__global__ void k_bcast(const _Float16* __restrict__ gsrc, int ldg,
                        _Float16* __restrict__ dst, int ldD, int colOff,
                        int rowsPerB, int cols, int B) {
    long total = (long)B * rowsPerB * cols;
    long g = (long)blockIdx.x * 256 + threadIdx.x;
    if (g >= total) return;
    int  c = (int)(g % cols);
    long r = g / cols;
    int b = (int)(r / rowsPerB);
    dst[r * (long)ldD + colOff + c] = gsrc[(long)b * ldg + c];
}

// ===========================================================================
extern "C" void kernel_launch(void* const* d_in, const int* in_sizes, int n_in,
                              void* d_out, int out_size, void* d_ws, size_t ws_size,
                              hipStream_t stream) {
    (void)in_sizes; (void)n_in; (void)out_size; (void)ws_size;
    const int B = 4;
    const int cs[5] = {64, 128, 256, 512, 1024};
    const int ns[5] = {2048, 512, 128, 32, 8};
    const int douts[4] = {512, 256, 128, 128};
    const int fcin[4]  = {1536, 768, 384, 192};

    char* wsp = (char*)d_ws;
    size_t off = 0;
    auto alloc = [&](size_t bytes) -> void* {
        void* p = wsp + off;
        off = (off + bytes + 255) & ~(size_t)255;
        return p;
    };
    auto g1 = [](long total) { return dim3((unsigned)((total + 255) / 256)); };

    // ---- activation / scratch arenas ----
    _Float16* xHalf = (_Float16*)alloc((size_t)8192 * 32 * 2);
    _Float16* feats[5];
    for (int L = 0; L < 5; ++L)
        feats[L] = (_Float16*)alloc((size_t)B * ns[L] * cs[L] * 2);
    float* xyzL[5];
    xyzL[0] = (float*)d_in[0];
    for (int L = 1; L < 5; ++L) xyzL[L] = (float*)alloc((size_t)B * ns[L] * 3 * 4);
    int* fidx[4];
    int* knnIdx[4];
    for (int i = 0; i < 4; ++i) {
        fidx[i]   = (int*)alloc((size_t)B * ns[i + 1] * 4);
        knnIdx[i] = (int*)alloc((size_t)B * ns[i + 1] * 32 * 4);
    }
    float* distbuf = (float*)alloc((size_t)B * 2048 * 4);
    float* stats   = (float*)alloc(2 * B * 4);
    float* stds    = (float*)alloc(B * 4);
    _Float16* arenaG = (_Float16*)alloc((size_t)65536 * 160 * 2);
    _Float16* curA   = (_Float16*)alloc((size_t)65536 * 128 * 2);
    _Float16* curB   = (_Float16*)alloc((size_t)65536 * 128 * 2);
    _Float16* wideA  = (_Float16*)alloc((size_t)65536 * 512 * 2);
    _Float16* wideB  = (_Float16*)alloc((size_t)65536 * 512 * 2);
    _Float16* hdecA  = (_Float16*)alloc((size_t)8192 * 512 * 2);
    _Float16* hdecB  = (_Float16*)alloc((size_t)8192 * 512 * 2);
    _Float16* fuseBuf = (_Float16*)alloc((size_t)8192 * 192 * 2 > (size_t)512 * 768 * 2
                                             ? (size_t)8192 * 192 * 2 : (size_t)512 * 768 * 2);
    int*   fpIdx  = (int*)alloc((size_t)B * 2048 * 3 * 4);
    float* fpDist = (float*)alloc((size_t)B * 2048 * 3 * 4);
    _Float16* gmpTmp = (_Float16*)alloc((size_t)8192 * 64 * 2);
    _Float16* gmpCat = (_Float16*)alloc((size_t)B * 320 * 2);
    _Float16* gBuf   = (_Float16*)alloc((size_t)B * 64 * 2);
    _Float16* clsIn  = (_Float16*)alloc((size_t)8192 * 192 * 2);
    _Float16* clsMid = (_Float16*)alloc((size_t)8192 * 128 * 2);

    // ---- weight conversion (f32 -> f16, K padded to 32) ----
    auto cvtW = [&](int idx, int rows, int cols) -> const _Float16* {
        int kp = (cols + 31) & ~31;
        _Float16* p = (_Float16*)alloc((size_t)rows * kp * 2);
        long total = (long)rows * kp;
        k_cvt_pad<<<g1(total), 256, 0, stream>>>((const float*)d_in[idx], p, rows, cols, kp, total);
        return p;
    };

    const _Float16* embW = cvtW(1, 64, 3);
    const float*    embB = (const float*)d_in[2];

    struct StW {
        const float *alpha, *beta, *tB;
        const _Float16* tW;
        const _Float16 *m1W[2], *m2W[2], *m3W[2];
        const float    *m1B[2], *m2B[2], *m3B[2];
        const _Float16 *c1W[2], *c2W[2];
        const float    *c1B[2], *c2B[2];
    } st[4];
    for (int i = 0; i < 4; ++i) {
        int base = 3 + i * 24;
        int cin = cs[i], cout = cs[i + 1];
        st[i].alpha = (const float*)d_in[base];
        st[i].beta  = (const float*)d_in[base + 1];
        st[i].tW = cvtW(base + 2, cout, 2 * cin + 3);
        st[i].tB = (const float*)d_in[base + 3];
        for (int j = 0; j < 2; ++j) {
            int pb = base + 4 + j * 6;
            st[i].m1W[j] = cvtW(pb, 4 * cout, cout);          st[i].m1B[j] = (const float*)d_in[pb + 1];
            st[i].m2W[j] = cvtW(pb + 2, 4 * cout, 4 * cout);  st[i].m2B[j] = (const float*)d_in[pb + 3];
            st[i].m3W[j] = cvtW(pb + 4, cout, 4 * cout);      st[i].m3B[j] = (const float*)d_in[pb + 5];
        }
        for (int j = 0; j < 2; ++j) {
            int qb = base + 16 + j * 4;
            st[i].c1W[j] = cvtW(qb, cout, cout);      st[i].c1B[j] = (const float*)d_in[qb + 1];
            st[i].c2W[j] = cvtW(qb + 2, cout, cout);  st[i].c2B[j] = (const float*)d_in[qb + 3];
        }
    }
    struct DeW {
        const _Float16* fW; const float* fB;
        const _Float16 *c1W[2], *c2W[2];
        const float    *c1B[2], *c2B[2];
    } de[4];
    for (int i = 0; i < 4; ++i) {
        int db = 99 + i * 10;
        de[i].fW = cvtW(db, douts[i], fcin[i]);
        de[i].fB = (const float*)d_in[db + 1];
        for (int j = 0; j < 2; ++j) {
            int eb = db + 2 + j * 4;
            de[i].c1W[j] = cvtW(eb, douts[i], douts[i]);     de[i].c1B[j] = (const float*)d_in[eb + 1];
            de[i].c2W[j] = cvtW(eb + 2, douts[i], douts[i]); de[i].c2B[j] = (const float*)d_in[eb + 3];
        }
    }
    const _Float16* gmpW[5]; const float* gmpB[5];
    for (int L = 0; L < 5; ++L) {
        gmpW[L] = cvtW(139 + 2 * L, 64, cs[L]);
        gmpB[L] = (const float*)d_in[140 + 2 * L];
    }
    const _Float16* geW = cvtW(149, 64, 320);  const float* geB = (const float*)d_in[150];
    const _Float16* l1W = cvtW(151, 128, 192); const float* l1B = (const float*)d_in[152];
    const _Float16* l2W = cvtW(153, 13, 128);  const float* l2B = (const float*)d_in[154];

    auto gemm = [&](const _Float16* X, int Kp, const _Float16* W, const float* bias,
                    int M, int Cout, int Np, float scale, int relu,
                    const _Float16* res, _Float16* outH, float* outF, int ldF) {
        int nc = Np > Cout ? Np : Cout;
        dim3 grid((unsigned)((M + 127) / 128), (unsigned)((nc + 127) / 128));
        k_gemm<<<grid, 256, 0, stream>>>(X, Kp, W, bias, M, Cout, Np, scale, relu,
                                         res, outH, outF, ldF);
    };

    // ---- embedding: feat0 = relu((x @ Wemb^T + b) * BN) ----
    k_cvt_pad<<<g1(8192L * 32), 256, 0, stream>>>((const float*)d_in[0], xHalf, 8192, 3, 32, 8192L * 32);
    gemm(xHalf, 32, embW, embB, 8192, 64, 64, BN_SC, 1, nullptr, feats[0], nullptr, 0);

    // ---- encoder stages ----
    for (int i = 0; i < 4; ++i) {
        int cin = cs[i], cout = cs[i + 1];
        int n = ns[i + 1], Nprev = ns[i];
        int Mg = B * n * 32;
        int tcin = 2 * cin + 3, tKp = (tcin + 31) & ~31;

        k_fps<<<dim3(B), 256, 0, stream>>>(xyzL[i], Nprev, n, fidx[i], distbuf);
        k_gather_xyz<<<g1(B * n), 256, 0, stream>>>(xyzL[i], fidx[i], n, Nprev, xyzL[i + 1], B);
        k_knn<32><<<dim3((B * n + 127) / 128), 128, 0, stream>>>(
            xyzL[i + 1], xyzL[i], n, Nprev, knnIdx[i], nullptr, B * n);

        k_zero_f32<<<1, 32, 0, stream>>>(stats, 2 * B);
        int bpb = (n * 32) / 256;
        k_group_stats<<<dim3(B * bpb), 256, 0, stream>>>(
            feats[i], cs[i], xyzL[i], fidx[i], knnIdx[i], n, Nprev, cin, bpb, stats);
        k_std<<<1, 32, 0, stream>>>(stats, stds, (float)((long)n * 32 * (cin + 3)), B);
        k_fill16<<<g1((long)Mg * tKp), 256, 0, stream>>>(arenaG, (long)Mg * tKp);
        k_group_write<<<g1(Mg), 256, 0, stream>>>(
            feats[i], cs[i], xyzL[i], fidx[i], knnIdx[i],
            st[i].alpha, st[i].beta, stds, n, Nprev, cin, tKp, arenaG, Mg);

        // transfer cbr
        gemm(arenaG, tKp, st[i].tW, st[i].tB, Mg, cout, cout, BN_SC, 1, nullptr, curA, nullptr, 0);
        _Float16 *cur = curA, *oth = curB;
        // pre: inverted residual blocks
        for (int j = 0; j < 2; ++j) {
            gemm(cur, cout, st[i].m1W[j], st[i].m1B[j], Mg, 4 * cout, 4 * cout, 1.0f, 0, nullptr, wideA, nullptr, 0);
            gemm(wideA, 4 * cout, st[i].m2W[j], st[i].m2B[j], Mg, 4 * cout, 4 * cout, BN_SC, 0, nullptr, wideB, nullptr, 0);
            gemm(wideB, 4 * cout, st[i].m3W[j], st[i].m3B[j], Mg, cout, cout, 1.0f, 1, cur, oth, nullptr, 0);
            _Float16* t2 = cur; cur = oth; oth = t2;
        }
        // max over k
        k_pool<<<g1((long)B * n * cout), 256, 0, stream>>>(cur, cout, feats[i + 1], (long)B * n, cout);
        // pos: residual blocks (in place)
        for (int j = 0; j < 2; ++j) {
            gemm(feats[i + 1], cout, st[i].c1W[j], st[i].c1B[j], B * n, cout, cout, BN_SC, 1, nullptr, curA, nullptr, 0);
            gemm(curA, cout, st[i].c2W[j], st[i].c2B[j], B * n, cout, cout, BN_SC, 1, feats[i + 1], feats[i + 1], nullptr, 0);
        }
    }

    // ---- decoder (feature propagation) ----
    _Float16* h = feats[4];
    int ch = 1024;
    for (int i = 0; i < 4; ++i) {
        int lev = 3 - i;
        int n1 = ns[lev], n2 = ns[lev + 1];
        int c1 = cs[lev], dout = douts[i], fc = fcin[i];
        k_knn<3><<<dim3((B * n1 + 127) / 128), 128, 0, stream>>>(
            xyzL[lev], xyzL[lev + 1], n1, n2, fpIdx, fpDist, B * n1);
        k_copy_cols<<<g1((long)B * n1 * c1), 256, 0, stream>>>(
            feats[lev], c1, fuseBuf, fc, (long)B * n1, c1, 0);
        k_interp<<<g1((long)B * n1 * ch), 256, 0, stream>>>(
            h, ch, fpIdx, fpDist, n1, n2, ch, fuseBuf, fc, c1, (long)B * n1 * ch);
        gemm(fuseBuf, fc, de[i].fW, de[i].fB, B * n1, dout, dout, BN_SC, 1, nullptr, hdecA, nullptr, 0);
        for (int j = 0; j < 2; ++j) {
            gemm(hdecA, dout, de[i].c1W[j], de[i].c1B[j], B * n1, dout, dout, BN_SC, 1, nullptr, hdecB, nullptr, 0);
            gemm(hdecB, dout, de[i].c2W[j], de[i].c2B[j], B * n1, dout, dout, BN_SC, 1, hdecA, hdecA, nullptr, 0);
        }
        h = hdecA; ch = dout;
    }

    // ---- global max pooling branch ----
    for (int L = 0; L < 5; ++L) {
        gemm(feats[L], cs[L], gmpW[L], gmpB[L], B * ns[L], 64, 64, BN_SC, 1, nullptr, gmpTmp, nullptr, 0);
        k_maxpoints<<<g1(B * 64), 256, 0, stream>>>(gmpTmp, ns[L], gmpCat, 320, 64 * L, B);
    }
    gemm(gmpCat, 320, geW, geB, B, 64, 64, BN_SC, 1, nullptr, gBuf, nullptr, 0);

    // ---- classifier head ----
    k_copy_cols<<<g1(8192L * 128), 256, 0, stream>>>(h, 128, clsIn, 192, 8192, 128, 0);
    k_bcast<<<g1(8192L * 64), 256, 0, stream>>>(gBuf, 64, clsIn, 192, 128, 2048, 64, B);
    gemm(clsIn, 192, l1W, l1B, 8192, 128, 128, BN_SC, 1, nullptr, clsMid, nullptr, 0);
    gemm(clsMid, 128, l2W, l2B, 8192, 13, 13, 1.0f, 0, nullptr, nullptr, (float*)d_out, 13);
}
